// AsrModel_73684458930746
// MI455X (gfx1250) — matrix operand order repaired and checked by reading, not compiled, not run on hardware
//
#include <hip/hip_runtime.h>
#include <hip/hip_bf16.h>

typedef __attribute__((ext_vector_type(16))) __bf16 v16bf;
typedef __attribute__((ext_vector_type(8)))  __bf16 v8bf;
typedef __attribute__((ext_vector_type(8)))  float  v8f;

namespace {
constexpr int T_ = 500, B_ = 1024, E_ = 26, H_ = 100, P_ = 80, D_ = 64;
constexpr int KPE = 32;    // E padded to one bf16 WMMA K-slab
constexpr int KPH = 128;   // H padded to four K-slabs
constexpr int NP  = 112;   // H padded in the N dimension (7 tiles)
constexpr int NT  = 7;
constexpr int MT  = B_ / 16;        // 64
constexpr int NWAVE = MT * NT;      // 448 waves exactly

constexpr size_t alignUp(size_t x) { return (x + 255) & ~size_t(255); }

// ---- workspace layout (bytes) ----
constexpr size_t off_bar   = 0;                                  // 64 u32 barrier counters
constexpr size_t off_xenc  = 256;                                // bf16 [T][B][KPE]
constexpr size_t sz_xenc   = (size_t)T_ * B_ * KPE * 2;
constexpr size_t off_ys0   = alignUp(off_xenc + sz_xenc);        // bf16 [T][B][KPH] (enc0 outputs)
constexpr size_t sz_ysT    = (size_t)T_ * B_ * KPH * 2;
constexpr size_t off_decin = alignUp(off_ys0 + sz_ysT);          // bf16 [P][B][KPE]
constexpr size_t sz_decin  = (size_t)P_ * B_ * KPE * 2;
constexpr size_t off_d0    = alignUp(off_decin + sz_decin);      // bf16 [P][B][KPH]
constexpr size_t sz_ysP    = (size_t)P_ * B_ * KPH * 2;
constexpr size_t off_d1    = alignUp(off_d0 + sz_ysP);           // bf16 [P][B][KPH]
constexpr size_t off_ring  = alignUp(off_d1 + sz_ysP);           // bf16 [2][B][KPH]
constexpr size_t sz_ring   = (size_t)2 * B_ * KPH * 2;
constexpr size_t off_henc0 = alignUp(off_ring + sz_ring);        // bf16 [B][KPH]
constexpr size_t sz_h      = (size_t)B_ * KPH * 2;
constexpr size_t off_henc1 = alignUp(off_henc0 + sz_h);
constexpr size_t sz_wE     = (size_t)3 * NP * KPE * 2;           // 21504
constexpr size_t sz_wH     = (size_t)3 * NP * KPH * 2;           // 86016
constexpr size_t off_wih_e0 = alignUp(off_henc1 + sz_h);
constexpr size_t off_whh_e0 = alignUp(off_wih_e0 + sz_wE);
constexpr size_t off_wih_e1 = alignUp(off_whh_e0 + sz_wH);
constexpr size_t off_whh_e1 = alignUp(off_wih_e1 + sz_wH);
constexpr size_t off_wih_d0 = alignUp(off_whh_e1 + sz_wH);
constexpr size_t off_whh_d0 = alignUp(off_wih_d0 + sz_wE);
constexpr size_t off_wih_d1 = alignUp(off_whh_d0 + sz_wH);
constexpr size_t off_whh_d1 = alignUp(off_wih_d1 + sz_wH);
constexpr size_t off_wlin   = alignUp(off_whh_d1 + sz_wH);       // bf16 [D][KPH]

// d_out layout: softmax_cal | target_cal | asr_outputs
constexpr size_t out_sm   = (size_t)(P_ - 1) * B_ * D_;  // 5,177,344
constexpr size_t out_rows = (size_t)(P_ - 1) * B_;       // 80,896
} // namespace

// ------------------------------------------------------------------ helpers

union AFrag { v16bf v; v8bf h[2]; };

__device__ __forceinline__ v8f wmma_bf16(v16bf a, v16bf b, v8f c) {
  return __builtin_amdgcn_wmma_f32_16x16x32_bf16(false, a, false, b, (short)0, c,
                                                 false, false);
}

__device__ __forceinline__ void grid_sync(unsigned* bar, unsigned expected) {
  __syncthreads();
  if (threadIdx.x == 0) {
    __threadfence();  // publish this block's writes device-wide
    __hip_atomic_fetch_add(bar, 1u, __ATOMIC_RELEASE, __HIP_MEMORY_SCOPE_AGENT);
    while (__hip_atomic_load(bar, __ATOMIC_ACQUIRE, __HIP_MEMORY_SCOPE_AGENT) < expected)
      __builtin_amdgcn_s_sleep(1);
  }
  __syncthreads();
  __threadfence();
}

// ------------------------------------------------------------------ prep kernels

__global__ void k_zero_bar(unsigned* bar) {
  if (threadIdx.x < 64) bar[threadIdx.x] = 0u;
}

__global__ void k_cvt_x(const float* __restrict__ x, __bf16* __restrict__ out, long n) {
  long i = (long)blockIdx.x * blockDim.x + threadIdx.x;
  if (i >= n) return;
  int k = (int)(i % KPE);
  long row = i / KPE;
  float v = (k < E_) ? x[row * E_ + k] : 0.f;
  out[i] = (__bf16)v;
}

__global__ void k_embed(const int* __restrict__ tgt, const float* __restrict__ emb,
                        __bf16* __restrict__ out, long n) {
  long i = (long)blockIdx.x * blockDim.x + threadIdx.x;
  if (i >= n) return;
  int k = (int)(i % KPE);
  long row = i / KPE;
  int t = tgt[row];
  float v = (k < E_) ? emb[t * E_ + k] : 0.f;
  out[i] = (__bf16)v;
}

// W fp32 [3*H][IN] -> bf16 [3][NP][KP], zero padded
__global__ void k_pack_gru_w(const float* __restrict__ W, int IN, int KP,
                             __bf16* __restrict__ out) {
  int i = blockIdx.x * blockDim.x + threadIdx.x;
  int tot = 3 * NP * KP;
  if (i >= tot) return;
  int k = i % KP;
  int r = (i / KP) % NP;
  int g = i / (KP * NP);
  float v = (r < H_ && k < IN) ? W[(g * H_ + r) * IN + k] : 0.f;
  out[i] = (__bf16)v;
}

__global__ void k_pack_lin(const float* __restrict__ W, __bf16* __restrict__ out) {
  int i = blockIdx.x * blockDim.x + threadIdx.x;
  if (i >= D_ * KPH) return;
  int k = i % KPH;
  int d = i / KPH;
  float v = (k < H_) ? W[d * H_ + k] : 0.f;
  out[i] = (__bf16)v;
}

// ------------------------------------------------------------------ persistent GRU

// Each wave owns one 16x16 (rows of B, cols of H) tile of the hidden state and
// all three gates for it.  Recurrent weights (Whh) stay pinned in VGPRs for the
// whole time loop (they are on the serial critical path).  Input-projection
// weights (Wih): tiny for IKS==1 -> registers; for IKS==4 they are staged once
// into LDS and re-fetched per step via ds_load (off the critical path), which
// keeps VGPR pressure under the 256 budget (no scratch spills in the loop).
// One grid barrier per step; hidden state double-buffered (bf16 ring, K padded).
template <int IKS>
__global__ __launch_bounds__(128, 1) void k_gru(
    const __bf16* __restrict__ xin,  // [nsteps][B][IKS*32]
    const __bf16* __restrict__ wih,  // [3][NP][IKS*32]
    const __bf16* __restrict__ whh,  // [3][NP][KPH]
    const float* __restrict__ bih, const float* __restrict__ bhh,  // [3*H]
    const __bf16* __restrict__ h_init,  // [B][KPH] or null (zeros)
    __bf16* __restrict__ ring,          // [2][B][KPH]
    __bf16* __restrict__ ys_out,        // [nsteps][B][KPH] or null
    __bf16* __restrict__ h_final,       // [B][KPH] or null
    int nsteps, unsigned* bar) {
  constexpr int KIN = IKS * 32;
  extern __shared__ __bf16 s_wih[];  // [3][NP][KIN] for IKS==4
  const int lane = threadIdx.x & 31;
  const int wid = blockIdx.x * (blockDim.x >> 5) + (threadIdx.x >> 5);  // 0..447
  const int mt = wid / NT, nt = wid % NT;
  const int row0 = mt * 16, col0 = nt * 16;
  const int lh = lane >> 4, l15 = lane & 15;
  const unsigned nb = gridDim.x;
  unsigned ep = 0;

  // Recurrent B fragments: column N = l15, K = lh*16 + e -> one 32B load per tile
  v16bf bh[3][4];
#pragma unroll
  for (int g = 0; g < 3; ++g)
#pragma unroll
    for (int ks = 0; ks < 4; ++ks)
      bh[g][ks] = *(const v16bf*)(whh + ((size_t)(g * NP + col0 + l15) * KPH + ks * 32 + lh * 16));

  v16bf bi[3][IKS];  // only register-resident for IKS==1
  if constexpr (IKS == 1) {
#pragma unroll
    for (int g = 0; g < 3; ++g)
      bi[g][0] = *(const v16bf*)(wih + ((size_t)(g * NP + col0 + l15) * KIN + lh * 16));
  } else {
    // cooperative stage of Wih into LDS (16B per thread per iteration)
    constexpr int NV = 3 * NP * KIN / 8;  // #16-byte vectors
    const uint4* src = (const uint4*)wih;
    uint4* dst = (uint4*)s_wih;
    for (int i = threadIdx.x; i < NV; i += blockDim.x) dst[i] = src[i];
    __syncthreads();
  }
  // per-wave base into the LDS copy of Wih
  const __bf16* bi_lds = s_wih + (size_t)(col0 + l15) * KIN + lh * 16;

  const int bc = col0 + l15;
  const float b_r  = (bc < H_) ? bih[0 * H_ + bc] + bhh[0 * H_ + bc] : 0.f;
  const float b_z  = (bc < H_) ? bih[1 * H_ + bc] + bhh[1 * H_ + bc] : 0.f;
  const float b_in = (bc < H_) ? bih[2 * H_ + bc] : 0.f;
  const float b_hn = (bc < H_) ? bhh[2 * H_ + bc] : 0.f;

  // Init ring slot 0 (including zero K-pad columns 112..127 via nt==NT-1 waves).
#pragma unroll
  for (int v = 0; v < 8; ++v) {
    int row = row0 + v + lh * 8, col = col0 + l15;
    __bf16 hv = h_init ? h_init[(size_t)row * KPH + col] : (__bf16)0.f;
    ring[(size_t)row * KPH + col] = hv;
    if (nt == NT - 1) ring[(size_t)row * KPH + col + 16] = (__bf16)0.f;
  }
  grid_sync(bar, (++ep) * nb);

  for (int t = 0; t < nsteps; ++t) {
    const int rp = t & 1;
    const __bf16* hrd = ring + (size_t)rp * B_ * KPH;
    __bf16* hwr = ring + (size_t)(1 - rp) * B_ * KPH;

    v8f accr, accz, accin, acchn;
#pragma unroll
    for (int i = 0; i < 8; ++i) { accr[i] = b_r; accz[i] = b_z; accin[i] = b_in; acchn[i] = b_hn; }

    const int arow = row0 + l15;
    const __bf16* xr = xin + (size_t)t * B_ * KIN + (size_t)arow * KIN;
    if (t + 1 < nsteps)
      __builtin_prefetch(xin + (size_t)(t + 1) * B_ * KIN + (size_t)arow * KIN, 0, 1);

    // input projection (gi) fused into the same accumulators
#pragma unroll
    for (int ks = 0; ks < IKS; ++ks) {
      AFrag a;
      a.h[0] = *(const v8bf*)(xr + ks * 32 + lh * 16);
      a.h[1] = *(const v8bf*)(xr + ks * 32 + lh * 16 + 8);
      v16bf w0, w1, w2;
      if constexpr (IKS == 1) {
        w0 = bi[0][ks]; w1 = bi[1][ks]; w2 = bi[2][ks];
      } else {
        w0 = *(const v16bf*)(bi_lds + (size_t)(0 * NP) * KIN + ks * 32);
        w1 = *(const v16bf*)(bi_lds + (size_t)(1 * NP) * KIN + ks * 32);
        w2 = *(const v16bf*)(bi_lds + (size_t)(2 * NP) * KIN + ks * 32);
      }
      accr  = wmma_bf16(a.v, w0, accr);
      accz  = wmma_bf16(a.v, w1, accz);
      accin = wmma_bf16(a.v, w2, accin);
    }
    // recurrent projection (gh) — register-resident weights
    const __bf16* hr = hrd + (size_t)arow * KPH;
#pragma unroll
    for (int ks = 0; ks < 4; ++ks) {
      AFrag a;
      a.h[0] = *(const v8bf*)(hr + ks * 32 + lh * 16);
      a.h[1] = *(const v8bf*)(hr + ks * 32 + lh * 16 + 8);
      accr  = wmma_bf16(a.v, bh[0][ks], accr);
      accz  = wmma_bf16(a.v, bh[1][ks], accz);
      acchn = wmma_bf16(a.v, bh[2][ks], acchn);
    }

    __bf16* yr = ys_out ? (ys_out + (size_t)t * B_ * KPH) : nullptr;
    const bool last = (t == nsteps - 1);
#pragma unroll
    for (int v = 0; v < 8; ++v) {
      int row = row0 + v + lh * 8, col = col0 + l15;
      float r  = 1.f / (1.f + __expf(-accr[v]));
      float z  = 1.f / (1.f + __expf(-accz[v]));
      float nn = tanhf(accin[v] + r * acchn[v]);
      float hp = (float)hrd[(size_t)row * KPH + col];
      float hv = (col < H_) ? ((1.f - z) * nn + z * hp) : 0.f;
      __bf16 hb = (__bf16)hv;
      hwr[(size_t)row * KPH + col] = hb;
      if (nt == NT - 1) hwr[(size_t)row * KPH + col + 16] = (__bf16)0.f;
      if (yr) {
        yr[(size_t)row * KPH + col] = hb;
        if (nt == NT - 1) yr[(size_t)row * KPH + col + 16] = (__bf16)0.f;
      }
      if (h_final && last) {
        h_final[(size_t)row * KPH + col] = hb;
        if (nt == NT - 1) h_final[(size_t)row * KPH + col + 16] = (__bf16)0.f;
      }
    }
    grid_sync(bar, (++ep) * nb);
  }
}

// ------------------------------------------------------------------ final linear (WMMA)

// logits[r, d] = d1[r, :H] @ lin_W[d, :H] + lin_b[d], r in [0, 79*1024)
__global__ __launch_bounds__(256, 1) void k_linear(
    const __bf16* __restrict__ d1, const __bf16* __restrict__ wlin,
    const float* __restrict__ linb, float* __restrict__ out) {
  const int lane = threadIdx.x & 31;
  const int wid = blockIdx.x * (blockDim.x >> 5) + (threadIdx.x >> 5);
  const int m = wid >> 2, n = wid & 3;  // 5056 x 4 tiles
  const int lh = lane >> 4, l15 = lane & 15;
  const int col = n * 16 + l15;
  v8f acc;
  float b = linb[col];
#pragma unroll
  for (int i = 0; i < 8; ++i) acc[i] = b;
  const int arow = m * 16 + l15;
  const __bf16* ar = d1 + (size_t)arow * KPH;
#pragma unroll
  for (int ks = 0; ks < 4; ++ks) {
    AFrag a;
    a.h[0] = *(const v8bf*)(ar + ks * 32 + lh * 16);
    a.h[1] = *(const v8bf*)(ar + ks * 32 + lh * 16 + 8);
    v16bf bw = *(const v16bf*)(wlin + (size_t)col * KPH + ks * 32 + lh * 16);
    acc = wmma_bf16(a.v, bw, acc);
  }
#pragma unroll
  for (int v = 0; v < 8; ++v) {
    int row = m * 16 + v + lh * 8;
    out[(size_t)row * D_ + col] = acc[v];
  }
}

// argmax over D=64 logits per row (first-max wins, like jnp.argmax) + target copy
__global__ void k_finalize(const float* __restrict__ logits, const int* __restrict__ tgt,
                           float* __restrict__ out_tc, float* __restrict__ out_asr) {
  int r = blockIdx.x * blockDim.x + threadIdx.x;
  if (r >= (int)out_rows) return;
  const float* lr = logits + (size_t)r * D_;
  float mx = lr[0];
  int am = 0;
#pragma unroll 8
  for (int d = 1; d < D_; ++d) {
    float v = lr[d];
    if (v > mx) { mx = v; am = d; }
  }
  out_asr[r] = (float)am;
  out_tc[r] = (float)tgt[B_ + r];  // target[1:] flattened
}

// ------------------------------------------------------------------ launch

extern "C" void kernel_launch(void* const* d_in, const int* in_sizes, int n_in,
                              void* d_out, int out_size, void* d_ws, size_t ws_size,
                              hipStream_t stream) {
  (void)in_sizes; (void)n_in; (void)out_size; (void)ws_size;
  const float* x      = (const float*)d_in[0];
  const int*   target = (const int*)d_in[1];
  const float* emb    = (const float*)d_in[2];
  const float* eWih0  = (const float*)d_in[3];
  const float* eWhh0  = (const float*)d_in[4];
  const float* ebih0  = (const float*)d_in[5];
  const float* ebhh0  = (const float*)d_in[6];
  const float* eWih1  = (const float*)d_in[7];
  const float* eWhh1  = (const float*)d_in[8];
  const float* ebih1  = (const float*)d_in[9];
  const float* ebhh1  = (const float*)d_in[10];
  const float* dWih0  = (const float*)d_in[11];
  const float* dWhh0  = (const float*)d_in[12];
  const float* dbih0  = (const float*)d_in[13];
  const float* dbhh0  = (const float*)d_in[14];
  const float* dWih1  = (const float*)d_in[15];
  const float* dWhh1  = (const float*)d_in[16];
  const float* dbih1  = (const float*)d_in[17];
  const float* dbhh1  = (const float*)d_in[18];
  const float* linW   = (const float*)d_in[19];
  const float* linb   = (const float*)d_in[20];

  char* ws = (char*)d_ws;
  unsigned* bar   = (unsigned*)(ws + off_bar);
  __bf16* xenc    = (__bf16*)(ws + off_xenc);
  __bf16* ys0     = (__bf16*)(ws + off_ys0);
  __bf16* decin   = (__bf16*)(ws + off_decin);
  __bf16* d0buf   = (__bf16*)(ws + off_d0);
  __bf16* d1buf   = (__bf16*)(ws + off_d1);
  __bf16* ring    = (__bf16*)(ws + off_ring);
  __bf16* henc0   = (__bf16*)(ws + off_henc0);
  __bf16* henc1   = (__bf16*)(ws + off_henc1);
  __bf16* wih_e0  = (__bf16*)(ws + off_wih_e0);
  __bf16* whh_e0  = (__bf16*)(ws + off_whh_e0);
  __bf16* wih_e1  = (__bf16*)(ws + off_wih_e1);
  __bf16* whh_e1  = (__bf16*)(ws + off_whh_e1);
  __bf16* wih_d0  = (__bf16*)(ws + off_wih_d0);
  __bf16* whh_d0  = (__bf16*)(ws + off_whh_d0);
  __bf16* wih_d1  = (__bf16*)(ws + off_wih_d1);
  __bf16* whh_d1  = (__bf16*)(ws + off_whh_d1);
  __bf16* wlin    = (__bf16*)(ws + off_wlin);
  float*  out     = (float*)d_out;

  k_zero_bar<<<1, 64, 0, stream>>>(bar);

  { long n = (long)T_ * B_ * KPE;
    k_cvt_x<<<(unsigned)((n + 255) / 256), 256, 0, stream>>>(x, xenc, n); }
  { long n = (long)P_ * B_ * KPE;
    k_embed<<<(unsigned)((n + 255) / 256), 256, 0, stream>>>(target, emb, decin, n); }

  const int wE = 3 * NP * KPE, wH = 3 * NP * KPH;
  k_pack_gru_w<<<(wE + 255) / 256, 256, 0, stream>>>(eWih0, E_, KPE, wih_e0);
  k_pack_gru_w<<<(wH + 255) / 256, 256, 0, stream>>>(eWhh0, H_, KPH, whh_e0);
  k_pack_gru_w<<<(wH + 255) / 256, 256, 0, stream>>>(eWih1, H_, KPH, wih_e1);
  k_pack_gru_w<<<(wH + 255) / 256, 256, 0, stream>>>(eWhh1, H_, KPH, whh_e1);
  k_pack_gru_w<<<(wE + 255) / 256, 256, 0, stream>>>(dWih0, E_, KPE, wih_d0);
  k_pack_gru_w<<<(wH + 255) / 256, 256, 0, stream>>>(dWhh0, H_, KPH, whh_d0);
  k_pack_gru_w<<<(wH + 255) / 256, 256, 0, stream>>>(dWih1, H_, KPH, wih_d1);
  k_pack_gru_w<<<(wH + 255) / 256, 256, 0, stream>>>(dWhh1, H_, KPH, whh_d1);
  k_pack_lin<<<(D_ * KPH + 255) / 256, 256, 0, stream>>>(linW, wlin);

  const dim3 ggrid(NWAVE / 4), gblk(128);
  const size_t ldsH = (size_t)3 * NP * KPH * 2;  // Wih staging for IKS==4 (86 KB)
  // encoder layer 0: x(E) -> ys0, final h -> henc0
  k_gru<1><<<ggrid, gblk, 0, stream>>>(xenc, wih_e0, whh_e0, ebih0, ebhh0,
                                       nullptr, ring, ys0, henc0, T_, bar + 0);
  // encoder layer 1: ys0(H) -> (ys unused), final h -> henc1
  k_gru<4><<<ggrid, gblk, ldsH, stream>>>(ys0, wih_e1, whh_e1, ebih1, ebhh1,
                                          nullptr, ring, nullptr, henc1, T_, bar + 1);
  // decoder layer 0: emb(E), h0 = henc0 -> d0
  k_gru<1><<<ggrid, gblk, 0, stream>>>(decin, wih_d0, whh_d0, dbih0, dbhh0,
                                       henc0, ring, d0buf, nullptr, P_, bar + 2);
  // decoder layer 1: d0(H), h0 = henc1 -> d1
  k_gru<4><<<ggrid, gblk, ldsH, stream>>>(d0buf, wih_d1, whh_d1, dbih1, dbhh1,
                                          henc1, ring, d1buf, nullptr, P_, bar + 3);

  // logits for p=0..78 (79*1024 rows = 5056 M-tiles x 4 N-tiles, 8 waves/block)
  k_linear<<<(5056 * 4) / 8, 256, 0, stream>>>(d1buf, wlin, linb, out);

  k_finalize<<<((int)out_rows + 255) / 256, 256, 0, stream>>>(
      out, target, out + out_sm, out + out_sm + out_rows);
}
